// LLMBlock_66151086293120
// MI455X (gfx1250) — 8-bit matrix operand order repaired and checked by reading, compiled, not run on hardware
//
#include <hip/hip_runtime.h>
#include <hip/hip_bf16.h>
#include <math.h>

typedef signed char s8;
typedef unsigned int u32;
typedef __attribute__((ext_vector_type(4)))  int      v4i;
typedef __attribute__((ext_vector_type(8)))  int      v8i;
typedef __attribute__((ext_vector_type(8)))  float    v8f;
typedef __attribute__((ext_vector_type(16))) _Float16 v16h;

#define DIMC 2048
#define NHEAD 16
#define NKVH 4
#define HDIM 128
#define KVDC 512
#define MLPC 5461
#define MLPP 5504
#define SEQ  2048
#define BATCH 2
#define TOK  (BATCH*SEQ)
#define EPSV 1.1920929e-07f

#if defined(__AMDGCN__) && \
    __has_builtin(__builtin_amdgcn_global_load_async_to_lds_b128) && \
    __has_builtin(__builtin_amdgcn_s_wait_asynccnt)
#define USE_ASYNC_LDS 1
#else
#define USE_ASYNC_LDS 0
#endif

#if USE_ASYNC_LDS
typedef __attribute__((address_space(1))) v4i* gas4;
typedef __attribute__((address_space(3))) v4i* las4;
__device__ __forceinline__ gas4 to_gas(const void* p) {
  return (gas4)(unsigned long long)(size_t)p;
}
__device__ __forceinline__ las4 to_las(const void* p) {
  return (las4)(u32)(unsigned long long)(size_t)p;
}
#endif


__global__ void k_zero(float* p, int n) {
  int i = blockIdx.x * blockDim.x + threadIdx.x;
  if (i < n) p[i] = 0.f;
}

__global__ __launch_bounds__(256) void k_abssum(const float* __restrict__ w,
                                                float* acc, int n) {
  __shared__ float red[256];
  float s = 0.f;
  for (int i = blockIdx.x * blockDim.x + threadIdx.x; i < n;
       i += gridDim.x * blockDim.x)
    s += fabsf(w[i]);
  red[threadIdx.x] = s; __syncthreads();
  for (int st = 128; st > 0; st >>= 1) {
    if (threadIdx.x < st) red[threadIdx.x] += red[threadIdx.x + st];
    __syncthreads();
  }
  if (threadIdx.x == 0) atomicAdd(acc, red[0]);
}

__global__ void k_wfinalize(const float* acc, float* invsw, float inv_count) {
  invsw[0] = fmaxf(acc[0] * inv_count, 1e-5f);
}

__global__ __launch_bounds__(256) void k_wquant(const float* __restrict__ w,
                                                s8* __restrict__ wq,
                                                const float* invsw,
                                                int R, int C, int Rp, int Cp) {
  float scale = 1.f / invsw[0];
  int total = Rp * Cp;
  for (int i = blockIdx.x * blockDim.x + threadIdx.x; i < total;
       i += gridDim.x * blockDim.x) {
    int r = i / Cp, c = i - r * Cp;
    float q = 0.f;
    if (r < R && c < C)
      q = fminf(fmaxf(rintf(w[(size_t)r * C + c] * scale), -1.f), 1.f);
    wq[(size_t)r * Cp + c] = (s8)(int)q;
  }
}

__global__ __launch_bounds__(256) void k_rmsnorm_quant(
    const float* __restrict__ x, const float* __restrict__ w,
    s8* __restrict__ xq, float* __restrict__ invs) {
  __shared__ float red[256];
  __shared__ float hbuf[DIMC];
  const int row = blockIdx.x;
  const float* xr = x + (size_t)row * DIMC;
  float ss = 0.f;
  for (int c = threadIdx.x; c < DIMC; c += 256) { float v = xr[c]; ss += v * v; }
  red[threadIdx.x] = ss; __syncthreads();
  for (int st = 128; st > 0; st >>= 1) {
    if (threadIdx.x < st) red[threadIdx.x] += red[threadIdx.x + st];
    __syncthreads();
  }
  float rms = rsqrtf(red[0] / (float)DIMC + EPSV);
  float mx = 0.f;
  for (int c = threadIdx.x; c < DIMC; c += 256) {
    float h = xr[c] * rms * w[c];
    hbuf[c] = h;
    mx = fmaxf(mx, fabsf(h));
  }
  __syncthreads();
  red[threadIdx.x] = mx; __syncthreads();
  for (int st = 128; st > 0; st >>= 1) {
    if (threadIdx.x < st) red[threadIdx.x] = fmaxf(red[threadIdx.x], red[threadIdx.x + st]);
    __syncthreads();
  }
  float amax = fmaxf(red[0], 1e-5f);
  float scale = 127.f / amax;
  for (int c = threadIdx.x; c < DIMC; c += 256)
    xq[(size_t)row * DIMC + c] =
        (s8)(int)fminf(fmaxf(rintf(hbuf[c] * scale), -128.f), 127.f);
  if (threadIdx.x == 0) invs[row] = amax / 127.f;
}

__global__ __launch_bounds__(256) void k_rowquant(const float* __restrict__ x,
                                                  s8* __restrict__ xq,
                                                  float* __restrict__ invs,
                                                  int C, int Cp) {
  __shared__ float red[256];
  const int row = blockIdx.x;
  const float* xr = x + (size_t)row * Cp;
  float mx = 0.f;
  for (int c = threadIdx.x; c < C; c += 256) mx = fmaxf(mx, fabsf(xr[c]));
  red[threadIdx.x] = mx; __syncthreads();
  for (int st = 128; st > 0; st >>= 1) {
    if (threadIdx.x < st) red[threadIdx.x] = fmaxf(red[threadIdx.x], red[threadIdx.x + st]);
    __syncthreads();
  }
  float amax = fmaxf(red[0], 1e-5f);
  float scale = 127.f / amax;
  for (int c = threadIdx.x; c < Cp; c += 256) {
    float q = 0.f;
    if (c < C) q = fminf(fmaxf(rintf(xr[c] * scale), -128.f), 127.f);
    xq[(size_t)row * Cp + c] = (s8)(int)q;
  }
  if (threadIdx.x == 0) invs[row] = amax / 127.f;
}

__global__ __launch_bounds__(256) void k_gemm_i8(
    const s8* __restrict__ X, const s8* __restrict__ W,
    const float* __restrict__ xinvs, const float* __restrict__ winvs,
    const float* __restrict__ bias, const float* __restrict__ resid,
    const float* __restrict__ gmul, float* __restrict__ Yf,
    _Float16* __restrict__ Yh, int K, int Nout) {
  __shared__ u32 lx[128 * 16];
  __shared__ u32 lw[64 * 16];
  const int tid = threadIdx.x;
  const int lane = tid & 31, wave = tid >> 5;
  const int row0 = blockIdx.y * 128;
  const int col0 = blockIdx.x * 64;
  const int r = lane & 15, hi = lane >> 4;
  v8i acc[4] = {};
  for (int k0 = 0; k0 < K; k0 += 64) {
#if USE_ASYNC_LDS
#pragma unroll
    for (int i = 0; i < 2; ++i) {
      int idx = tid + i * 256;
      int rr = idx >> 2, cc = idx & 3;
      __builtin_amdgcn_global_load_async_to_lds_b128(
          to_gas(X + (size_t)(row0 + rr) * K + k0 + cc * 16),
          to_las(&lx[rr * 16 + cc * 4]), 0, 0);
    }
    {
      int rr = tid >> 2, cc = tid & 3;
      __builtin_amdgcn_global_load_async_to_lds_b128(
          to_gas(W + (size_t)(col0 + rr) * K + k0 + cc * 16),
          to_las(&lw[rr * 16 + cc * 4]), 0, 0);
    }
#else
#pragma unroll
    for (int i = 0; i < 2; ++i) {
      int idx = tid + i * 256;
      int rr = idx >> 2, cc = idx & 3;
      *(int4*)&lx[rr * 16 + cc * 4] =
          *(const int4*)(X + (size_t)(row0 + rr) * K + k0 + cc * 16);
    }
    {
      int rr = tid >> 2, cc = tid & 3;
      *(int4*)&lw[rr * 16 + cc * 4] =
          *(const int4*)(W + (size_t)(col0 + rr) * K + k0 + cc * 16);
    }
#endif
    if (k0 + 64 < K) {
      __builtin_prefetch(X + (size_t)(row0 + (tid >> 1)) * K + k0 + 64 + (tid & 1) * 32, 0, 1);
      __builtin_prefetch(W + (size_t)(col0 + (tid & 63)) * K + k0 + 64, 0, 1);
    }
#if USE_ASYNC_LDS
    __builtin_amdgcn_s_wait_asynccnt(0);
#endif
    __syncthreads();
    alignas(32) u32 ta[8];
#pragma unroll
    for (int j = 0; j < 8; ++j)
      ta[j] = lx[(wave * 16 + r) * 16 + ((j >> 1) << 2) + (j & 1) + hi * 2];
    v8i a = *(const v8i*)ta;
#pragma unroll
    for (int nt = 0; nt < 4; ++nt) {
      alignas(32) u32 tb[8];
#pragma unroll
      for (int j = 0; j < 8; ++j)
        tb[j] = lw[(nt * 16 + r) * 16 + ((j >> 1) << 2) + (j & 1) + hi * 2];
      v8i b = *(const v8i*)tb;
      acc[nt] = __builtin_amdgcn_wmma_i32_16x16x64_iu8(true, a, true, b,
                                                       acc[nt], false, false);
    }
    __syncthreads();
  }
  const float winv = winvs[0];
#pragma unroll
  for (int nt = 0; nt < 4; ++nt) {
    int col = col0 + nt * 16 + r;
#pragma unroll
    for (int rr = 0; rr < 8; ++rr) {
      int row = row0 + wave * 16 + rr + hi * 8;
      float y = (float)acc[nt][rr] * (xinvs[row] * winv);
      if (bias)  y += bias[col];
      if (gmul)  { float g = gmul[(size_t)row * Nout + col];
                   y *= g / (1.f + __expf(-g)); }
      if (resid) y += resid[(size_t)row * Nout + col];
      size_t o = (size_t)row * Nout + col;
      if (Yf) Yf[o] = y;
      if (Yh) Yh[o] = (_Float16)y;
    }
  }
}

__global__ void k_rope(_Float16* __restrict__ buf, const int* __restrict__ pos,
                       int heads, int total) {
  int i = blockIdx.x * blockDim.x + threadIdx.x;
  if (i >= total) return;
  int pr = i % (HDIM / 2);
  int h  = (i / (HDIM / 2)) % heads;
  int t  = i / (heads * (HDIM / 2));
  float p = (float)pos[t];
  float inv = __expf(-(float)(2 * pr) * (9.210340371976184f / (float)HDIM));
  float c = cosf(p * inv), s = sinf(p * inv);
  size_t o = (size_t)t * heads * HDIM + (size_t)h * HDIM + 2 * pr;
  float e = (float)buf[o], d = (float)buf[o + 1];
  buf[o]     = (_Float16)(e * c - d * s);
  buf[o + 1] = (_Float16)(d * c + e * s);
}

__global__ __launch_bounds__(256) void k_attn(const _Float16* __restrict__ qh,
                                              const _Float16* __restrict__ kh,
                                              const _Float16* __restrict__ vh,
                                              float* __restrict__ ao) {
  __shared__ _Float16 lk[32 * HDIM];
  __shared__ _Float16 lvt[HDIM * 32];
  __shared__ _Float16 lp[8][16 * 32];
  const int tid = threadIdx.x, lane = tid & 31, wave = tid >> 5;
  const int r = lane & 15, hi = lane >> 4;
  const int h = blockIdx.y, b = blockIdx.z, kvh = h >> 2;
  const int q0 = blockIdx.x * 128 + wave * 16;
  v16h qa[4];
  {
    const _Float16* qrow = qh + ((size_t)(b * SEQ + q0 + r)) * DIMC + h * HDIM;
#pragma unroll
    for (int c = 0; c < 4; ++c) {
      alignas(32) u32 t8[8];
#pragma unroll
      for (int j = 0; j < 8; ++j) {
        int kp = ((j & 3) << 1) + ((j >> 2) << 4) + hi * 8;
        t8[j] = *(const u32*)(qrow + c * 32 + kp);
      }
      qa[c] = *(const v16h*)t8;
    }
  }
  float m[8], l[8];
  v8f o[8] = {};
#pragma unroll
  for (int i2 = 0; i2 < 8; ++i2) { m[i2] = -INFINITY; l[i2] = 0.f; }
  const int kend = blockIdx.x * 128 + 128;
  const float scale = 0.08838834764831845f;
  for (int kc = 0; kc < kend; kc += 32) {
#pragma unroll
    for (int it = 0; it < 2; ++it) {
      int i = tid + it * 256;
      int kr = i >> 4, c8 = i & 15;
      const _Float16* ks = kh + ((size_t)(b * SEQ + kc + kr)) * KVDC + kvh * HDIM + c8 * 8;
#if USE_ASYNC_LDS
      __builtin_amdgcn_global_load_async_to_lds_b128(
          to_gas(ks), to_las(&lk[kr * HDIM + c8 * 8]), 0, 0);
#else
      *(int4*)&lk[kr * HDIM + c8 * 8] = *(const int4*)ks;
#endif
      const _Float16* vs = vh + ((size_t)(b * SEQ + kc + kr)) * KVDC + kvh * HDIM + c8 * 8;
#pragma unroll
      for (int e = 0; e < 8; ++e) lvt[(c8 * 8 + e) * 32 + kr] = vs[e];
    }
#if USE_ASYNC_LDS
    __builtin_amdgcn_s_wait_asynccnt(0);
#endif
    __syncthreads();
    v8f s[2] = {};
#pragma unroll
    for (int t2 = 0; t2 < 2; ++t2) {
#pragma unroll
      for (int c = 0; c < 4; ++c) {
        alignas(32) u32 t8[8];
#pragma unroll
        for (int j = 0; j < 8; ++j) {
          int kp = ((j & 3) << 1) + ((j >> 2) << 4) + hi * 8;
          t8[j] = *(const u32*)&lk[(t2 * 16 + r) * HDIM + c * 32 + kp];
        }
        v16h bfr = *(const v16h*)t8;
        s[t2] = __builtin_amdgcn_wmma_f32_16x16x32_f16(false, qa[c], false, bfr,
                                                       (short)0, s[t2], false, false);
      }
    }
    float fac[8];
#pragma unroll
    for (int rr = 0; rr < 8; ++rr) {
      int qidx = q0 + rr + hi * 8;
      float v0 = s[0][rr] * scale;
      float v1 = s[1][rr] * scale;
      if (kc + r > qidx)      v0 = -INFINITY;
      if (kc + 16 + r > qidx) v1 = -INFINITY;
      float mx = fmaxf(v0, v1);
#pragma unroll
      for (int mk = 1; mk < 16; mk <<= 1) mx = fmaxf(mx, __shfl_xor(mx, mk, 32));
      float mn = fmaxf(m[rr], mx);
      float e0 = __expf(v0 - mn);
      float e1 = __expf(v1 - mn);
      float rs = e0 + e1;
#pragma unroll
      for (int mk = 1; mk < 16; mk <<= 1) rs += __shfl_xor(rs, mk, 32);
      fac[rr] = __expf(m[rr] - mn);
      m[rr] = mn;
      l[rr] = l[rr] * fac[rr] + rs;
      lp[wave][(rr + hi * 8) * 32 + r]      = (_Float16)e0;
      lp[wave][(rr + hi * 8) * 32 + 16 + r] = (_Float16)e1;
    }
#pragma unroll
    for (int j = 0; j < 8; ++j)
#pragma unroll
      for (int rr = 0; rr < 8; ++rr) o[j][rr] *= fac[rr];
    __syncthreads();
    alignas(32) u32 tp[8];
#pragma unroll
    for (int j = 0; j < 8; ++j) {
      int kp = ((j & 3) << 1) + ((j >> 2) << 4) + hi * 8;
      tp[j] = *(const u32*)&lp[wave][r * 32 + kp];
    }
    v16h pa = *(const v16h*)tp;
#pragma unroll
    for (int ht = 0; ht < 8; ++ht) {
      alignas(32) u32 tb[8];
#pragma unroll
      for (int j = 0; j < 8; ++j) {
        int kp = ((j & 3) << 1) + ((j >> 2) << 4) + hi * 8;
        tb[j] = *(const u32*)&lvt[(ht * 16 + r) * 32 + kp];
      }
      v16h bv = *(const v16h*)tb;
      o[ht] = __builtin_amdgcn_wmma_f32_16x16x32_f16(false, pa, false, bv,
                                                     (short)0, o[ht], false, false);
    }
    __syncthreads();
  }
#pragma unroll
  for (int ht = 0; ht < 8; ++ht) {
    int col = h * HDIM + ht * 16 + r;
#pragma unroll
    for (int rr = 0; rr < 8; ++rr) {
      int row = b * SEQ + q0 + rr + hi * 8;
      ao[(size_t)row * DIMC + col] = o[ht][rr] / l[rr];
    }
  }
}

extern "C" void kernel_launch(void* const* d_in, const int* in_sizes, int n_in,
                              void* d_out, int out_size, void* d_ws, size_t ws_size,
                              hipStream_t stream) {
  (void)in_sizes; (void)n_in; (void)out_size; (void)ws_size;
  const float* x   = (const float*)d_in[0];
  const int*   pos = (const int*)d_in[1];
  const float* n1w = (const float*)d_in[2];
  const float* qw  = (const float*)d_in[3];
  const float* qb  = (const float*)d_in[4];
  const float* kw  = (const float*)d_in[5];
  const float* kb  = (const float*)d_in[6];
  const float* vw  = (const float*)d_in[7];
  const float* vb  = (const float*)d_in[8];
  const float* ow  = (const float*)d_in[9];
  const float* n2w = (const float*)d_in[10];
  const float* gw  = (const float*)d_in[11];
  const float* uw  = (const float*)d_in[12];
  const float* dw  = (const float*)d_in[13];
  float* out = (float*)d_out;

  char* ws = (char*)d_ws;
  size_t off = 0;
  auto alloc = [&](size_t n) { size_t o = off; off += (n + 255) & ~(size_t)255; return o; };
  s8* wq_q = (s8*)(ws + alloc((size_t)DIMC * DIMC));
  s8* wq_k = (s8*)(ws + alloc((size_t)KVDC * DIMC));
  s8* wq_v = (s8*)(ws + alloc((size_t)KVDC * DIMC));
  s8* wq_o = (s8*)(ws + alloc((size_t)DIMC * DIMC));
  s8* wq_g = (s8*)(ws + alloc((size_t)MLPP * DIMC));
  s8* wq_u = (s8*)(ws + alloc((size_t)MLPP * DIMC));
  s8* wq_d = (s8*)(ws + alloc((size_t)DIMC * MLPP));
  float* accs = (float*)(ws + alloc(16 * sizeof(float)));
  float* winv = (float*)(ws + alloc(16 * sizeof(float)));
  s8*    h1q   = (s8*)(ws + alloc((size_t)TOK * DIMC));
  float* invs1 = (float*)(ws + alloc((size_t)TOK * 4));
  _Float16* qhb = (_Float16*)(ws + alloc((size_t)TOK * DIMC * 2));
  _Float16* khb = (_Float16*)(ws + alloc((size_t)TOK * KVDC * 2));
  _Float16* vhb = (_Float16*)(ws + alloc((size_t)TOK * KVDC * 2));
  float* ao    = (float*)(ws + alloc((size_t)TOK * DIMC * 4));
  s8*    aoq   = (s8*)(ws + alloc((size_t)TOK * DIMC));
  float* invso = (float*)(ws + alloc((size_t)TOK * 4));
  float* x1    = (float*)(ws + alloc((size_t)TOK * DIMC * 4));
  s8*    h2q   = (s8*)(ws + alloc((size_t)TOK * DIMC));
  float* invs2 = (float*)(ws + alloc((size_t)TOK * 4));
  float* gbuf  = (float*)(ws + alloc((size_t)TOK * MLPP * 4));
  s8*    hq3   = (s8*)(ws + alloc((size_t)TOK * MLPP));
  float* invs3 = (float*)(ws + alloc((size_t)TOK * 4));

  k_zero<<<1, 32, 0, stream>>>(accs, 16);
  const int cntQ = DIMC * DIMC, cntK = KVDC * DIMC, cntM = MLPC * DIMC;
  k_abssum<<<512, 256, 0, stream>>>(qw, accs + 0, cntQ);
  k_abssum<<<512, 256, 0, stream>>>(kw, accs + 1, cntK);
  k_abssum<<<512, 256, 0, stream>>>(vw, accs + 2, cntK);
  k_abssum<<<512, 256, 0, stream>>>(ow, accs + 3, cntQ);
  k_abssum<<<512, 256, 0, stream>>>(gw, accs + 4, cntM);
  k_abssum<<<512, 256, 0, stream>>>(uw, accs + 5, cntM);
  k_abssum<<<512, 256, 0, stream>>>(dw, accs + 6, cntM);
  k_wfinalize<<<1, 1, 0, stream>>>(accs + 0, winv + 0, 1.f / (float)cntQ);
  k_wfinalize<<<1, 1, 0, stream>>>(accs + 1, winv + 1, 1.f / (float)cntK);
  k_wfinalize<<<1, 1, 0, stream>>>(accs + 2, winv + 2, 1.f / (float)cntK);
  k_wfinalize<<<1, 1, 0, stream>>>(accs + 3, winv + 3, 1.f / (float)cntQ);
  k_wfinalize<<<1, 1, 0, stream>>>(accs + 4, winv + 4, 1.f / (float)cntM);
  k_wfinalize<<<1, 1, 0, stream>>>(accs + 5, winv + 5, 1.f / (float)cntM);
  k_wfinalize<<<1, 1, 0, stream>>>(accs + 6, winv + 6, 1.f / (float)cntM);
  k_wquant<<<2048, 256, 0, stream>>>(qw, wq_q, winv + 0, DIMC, DIMC, DIMC, DIMC);
  k_wquant<<<2048, 256, 0, stream>>>(kw, wq_k, winv + 1, KVDC, DIMC, KVDC, DIMC);
  k_wquant<<<2048, 256, 0, stream>>>(vw, wq_v, winv + 2, KVDC, DIMC, KVDC, DIMC);
  k_wquant<<<2048, 256, 0, stream>>>(ow, wq_o, winv + 3, DIMC, DIMC, DIMC, DIMC);
  k_wquant<<<2048, 256, 0, stream>>>(gw, wq_g, winv + 4, MLPC, DIMC, MLPP, DIMC);
  k_wquant<<<2048, 256, 0, stream>>>(uw, wq_u, winv + 5, MLPC, DIMC, MLPP, DIMC);
  k_wquant<<<2048, 256, 0, stream>>>(dw, wq_d, winv + 6, DIMC, MLPC, DIMC, MLPP);

  k_rmsnorm_quant<<<TOK, 256, 0, stream>>>(x, n1w, h1q, invs1);
  k_gemm_i8<<<dim3(DIMC / 64, TOK / 128), 256, 0, stream>>>(
      h1q, wq_q, invs1, winv + 0, qb, nullptr, nullptr, nullptr, qhb, DIMC, DIMC);
  k_gemm_i8<<<dim3(KVDC / 64, TOK / 128), 256, 0, stream>>>(
      h1q, wq_k, invs1, winv + 1, kb, nullptr, nullptr, nullptr, khb, DIMC, KVDC);
  k_gemm_i8<<<dim3(KVDC / 64, TOK / 128), 256, 0, stream>>>(
      h1q, wq_v, invs1, winv + 2, vb, nullptr, nullptr, nullptr, vhb, DIMC, KVDC);
  {
    int tq = TOK * NHEAD * (HDIM / 2);
    int tk = TOK * NKVH * (HDIM / 2);
    k_rope<<<(tq + 255) / 256, 256, 0, stream>>>(qhb, pos, NHEAD, tq);
    k_rope<<<(tk + 255) / 256, 256, 0, stream>>>(khb, pos, NKVH, tk);
  }
  k_attn<<<dim3(SEQ / 128, NHEAD, BATCH), 256, 0, stream>>>(qhb, khb, vhb, ao);
  k_rowquant<<<TOK, 256, 0, stream>>>(ao, aoq, invso, DIMC, DIMC);
  k_gemm_i8<<<dim3(DIMC / 64, TOK / 128), 256, 0, stream>>>(
      aoq, wq_o, invso, winv + 3, nullptr, x, nullptr, x1, nullptr, DIMC, DIMC);

  k_rmsnorm_quant<<<TOK, 256, 0, stream>>>(x1, n2w, h2q, invs2);
  k_gemm_i8<<<dim3(MLPP / 64, TOK / 128), 256, 0, stream>>>(
      h2q, wq_g, invs2, winv + 4, nullptr, nullptr, nullptr, gbuf, nullptr, DIMC, MLPP);
  k_gemm_i8<<<dim3(MLPP / 64, TOK / 128), 256, 0, stream>>>(
      h2q, wq_u, invs2, winv + 5, nullptr, nullptr, gbuf, gbuf, nullptr, DIMC, MLPP);
  k_rowquant<<<TOK, 256, 0, stream>>>(gbuf, hq3, invs3, MLPC, MLPP);
  k_gemm_i8<<<dim3(DIMC / 64, TOK / 128), 256, 0, stream>>>(
      hq3, wq_d, invs3, winv + 6, nullptr, x1, nullptr, out, nullptr, MLPP, DIMC);
}
